// MyModel_42417097015327
// MI455X (gfx1250) — matrix operand order repaired and checked by reading, not compiled, not run on hardware
//
#include <hip/hip_runtime.h>
#include <hip/hip_bf16.h>

typedef __attribute__((ext_vector_type(16))) _Float16 v16h;
typedef __attribute__((ext_vector_type(8)))  _Float16 h8;
typedef __attribute__((ext_vector_type(8)))  float    v8f;
typedef __attribute__((ext_vector_type(4)))  float    f4;

#define NEGC (-9e8f)

__device__ __forceinline__ float f_lrelu(float x){ return x >= 0.f ? x : 0.01f * x; }
__device__ __forceinline__ float f_relu (float x){ return x > 0.f ? x : 0.f; }
__device__ __forceinline__ float f_elu  (float x){ return x > 0.f ? x : expm1f(x); }
__device__ __forceinline__ float f_sigm (float x){ return 1.f / (1.f + expf(-x)); }

__device__ __forceinline__ float warp_allreduce_sum(float v){
#pragma unroll
  for (int o = 16; o > 0; o >>= 1) v += __shfl_xor(v, o, 32);
  return v;
}

// ---------------- weight convert: f32 -> f16, N-major [Npad][Kpad] ----------
// mode 0: src row-major [K,N]      -> dst[n*Kpad+k] = src[k*N+n]
// mode 1: src [N,K] (GRU Wi/Wh)    -> dst[n*Kpad+k] = src[n*K+k] (plain copy)
__global__ void k_conv_w(const float* __restrict__ src, _Float16* __restrict__ dst,
                         int K, int N, int Kpad, int Npad, int mode)
{
  int idx = blockIdx.x * blockDim.x + threadIdx.x;
  int tot = Npad * Kpad;
  if (idx >= tot) return;
  int n = idx / Kpad, k = idx % Kpad;
  float v = 0.f;
  if (k < K && n < N) v = mode ? src[(size_t)n * K + k] : src[(size_t)k * N + n];
  dst[idx] = (_Float16)v;
}

// ---------------- WMMA GEMM: C = act(A_f32 @ W + rowscale[m]*bias[n]) -------
// W is f16, N-major [Npad][Kpad], Npad % 64 == 0, Kpad = ceil32(K).
// Contract: M % 64 == 0 (all call sites satisfy this).
// Tile 64x64, BK=32, 8 waves; each wave: two v_wmma_f32_16x16x32_f16 / K-step.
// B tile staged with CDNA5 async global->LDS copy (ASYNCcnt); A tile staged
// through VALU (needs f32->f16 conversion).
// act: 0 none, 1 lrelu, 2 relu, 3 elu.
__launch_bounds__(256)
__global__ void k_gemm(const float* __restrict__ A, int lda,
                       const _Float16* __restrict__ W, int kpw,
                       const float* __restrict__ bias,
                       const float* __restrict__ rowscale,
                       float* __restrict__ C, int ldc,
                       float* __restrict__ C2,
                       int M, int N, int K, int act)
{
  __shared__ _Float16 As[64 * 40];   // [m][k], stride 40 halves (16B-aligned rows)
  __shared__ _Float16 Bs[64 * 40];   // [n][k]

  const int tid  = threadIdx.x;
  const int lane = tid & 31;
  const int wv   = tid >> 5;
  const int wm   = wv & 3;
  const int wn   = wv >> 2;
  const int m0   = blockIdx.y * 64;
  const int n0   = blockIdx.x * 64;

  v8f acc0 = {}; v8f acc1 = {};

  const int nkt  = (K + 31) >> 5;
  const int kmax = K - 1;
  // A staging: thread owns row (tid>>2), k-octet (tid&3)*8 -> one h8 LDS store
  const int amr  = tid >> 2;          // 0..63
  const int akq  = (tid & 3) << 3;    // 0,8,16,24
  // fast path: rows 16B-aligned (lda%4==0 and base 16B-aligned)
  const bool al16 = ((lda & 3) == 0) && ((((size_t)A) & 15) == 0);
  const float* arow = A + (size_t)(m0 + amr) * lda;
  // B staging: thread owns column (tid&63), k-octet (tid>>6)*8
  const int bnn  = tid & 63;
  const int bkq  = (tid >> 6) << 3;
  // per-lane LDS destination byte address for the async copy (low 32 bits of
  // the generic LDS pointer are the LDS byte offset, ISA 10.2)
  const unsigned int bs_lds = (unsigned int)(size_t)&Bs[bnn * 40 + bkq];

  for (int kt = 0; kt < nkt; ++kt) {
    const int k0 = kt << 5;

    // ---- stage B tile: async global->LDS b128 copy (ASYNCcnt) -----------
    {
      const _Float16* gsrc = W + (size_t)(n0 + bnn) * kpw + (k0 + bkq);
      asm volatile("global_load_async_to_lds_b128 %0, %1, off"
                   :: "v"(bs_lds), "v"(gsrc)
                   : "memory");
    }

    // ---- stage A tile (f32 -> f16 through VGPRs) -------------------------
    {
      h8 av;
      const int kb = k0 + akq;
      if (al16 && (kb + 7 < K)) {
        // two unconditional global_load_b128
        f4 f0 = *(const f4*)(arow + kb);
        f4 f1 = *(const f4*)(arow + kb + 4);
#pragma unroll
        for (int j = 0; j < 4; ++j) { av[j] = (_Float16)f0[j]; av[4 + j] = (_Float16)f1[j]; }
      } else if (kb >= K) {
        // fully past K (tail tile): zeros, no loads
#pragma unroll
        for (int j = 0; j < 8; ++j) av[j] = (_Float16)0.f;
      } else {
        // two-phase clamped scalar loads: batch all loads, then convert
        float va[8];
#pragma unroll
        for (int j = 0; j < 8; ++j) {
          const int gk  = kb + j;
          const int gkc = gk < K ? gk : kmax;
          va[j] = arow[gkc];
        }
#pragma unroll
        for (int j = 0; j < 8; ++j)
          av[j] = (kb + j < K) ? (_Float16)va[j] : (_Float16)0.f;
      }
      *(h8*)&As[amr * 40 + akq] = av;
    }

    // my async B copies done -> then block-wide visibility via barrier
    asm volatile("s_wait_asynccnt 0x0" ::: "memory");
    __syncthreads();

    // A fragment: lane row M=lane&15; halves 0..7 -> K kb..kb+7,
    // halves 8..15 -> K 16+kb..16+kb+7 with kb = 8*(lane>>4).  (ISA 7.12.2)
    {
      const int mrow = lane & 15;
      const int kb   = (lane >> 4) << 3;
      const _Float16* ap = &As[(wm * 16 + mrow) * 40];
      h8 alo = *(const h8*)(ap + kb);
      h8 ahi = *(const h8*)(ap + 16 + kb);
      v16h af;
#pragma unroll
      for (int i = 0; i < 8; ++i) { af[i] = alo[i]; af[8 + i] = ahi[i]; }

      // B fragment: lane column N=lane&15; K = kh..kh+7, 16+kh..16+kh+7, kh = 8*(lane>>4)
      const int kh   = (lane >> 4) << 3;
      const int ncol = lane & 15;
      const _Float16* bp0 = &Bs[(wn * 32 + ncol) * 40 + kh];
      h8 b0a = *(const h8*)(bp0); h8 b0b = *(const h8*)(bp0 + 16);
      const _Float16* bp1 = &Bs[(wn * 32 + 16 + ncol) * 40 + kh];
      h8 b1a = *(const h8*)(bp1); h8 b1b = *(const h8*)(bp1 + 16);
      v16h bf0, bf1;
#pragma unroll
      for (int i = 0; i < 8; ++i) {
        bf0[i] = b0a[i]; bf0[8 + i] = b0b[i];
        bf1[i] = b1a[i]; bf1[8 + i] = b1b[i];
      }
      acc0 = __builtin_amdgcn_wmma_f32_16x16x32_f16(false, af, false, bf0,
                                                    (short)0, acc0, false, false);
      acc1 = __builtin_amdgcn_wmma_f32_16x16x32_f16(false, af, false, bf1,
                                                    (short)0, acc1, false, false);
    }
    __syncthreads();
  }

  // epilogue: C-frag element r -> (m = r + 8*(lane>>4), n = lane&15)
  const int mbase = m0 + wm * 16 + ((lane >> 4) << 3);
#pragma unroll
  for (int r = 0; r < 8; ++r) {
    const int gm = mbase + r;                  // M % 64 == 0: always < M
    const float rsc = rowscale ? rowscale[gm] : 1.f;
#pragma unroll
    for (int t = 0; t < 2; ++t) {
      const int gn = n0 + wn * 32 + t * 16 + (lane & 15);
      if (gn >= N) continue;
      float v = (t == 0) ? acc0[r] : acc1[r];
      if (bias) v += rsc * bias[gn];
      if      (act == 1) v = f_lrelu(v);
      else if (act == 2) v = f_relu(v);
      else if (act == 3) v = f_elu(v);
      C[(size_t)gm * ldc + gn] = v;
      if (C2) C2[(size_t)gm * ldc + gn] = v;
    }
  }
}

// ---------------- neighbor attention round (one wave32 per atom) -------------
// first=1: nb_k = lrelu(P[idxA]+Q[idxB]+nbias)   (factored neighbor_fc)
// first=0: nb_k = cur[idxA]
// score_k = lrelu(cur.w_top + nb_k.w_bot + b) + mask ; softmax over K=6
// out: wsum = sum_k attn_k*nb_k   asum = sum_k attn_k
__launch_bounds__(256)
__global__ void k_neighbor(const float* __restrict__ cur,
                           const float* __restrict__ P, const float* __restrict__ Qb,
                           const float* __restrict__ nbb,
                           const int* __restrict__ idxA, const int* __restrict__ idxB,
                           const float* __restrict__ atop, const float* __restrict__ abot,
                           const float* __restrict__ ab,
                           float* __restrict__ wsum, float* __restrict__ asum, int first)
{
  const int lane = threadIdx.x & 31;
  const int atom = blockIdx.x * 8 + (threadIdx.x >> 5);
  const int b    = atom >> 7;

  float botv[7];
  float part = 0.f;
#pragma unroll
  for (int j = 0; j < 7; ++j) {
    const int f  = lane + 32 * j;
    const int fc = f < 200 ? f : 199;          // clamp, select later
    const float w  = (f < 200) ? 1.f : 0.f;
    const float cv = cur[(size_t)atom * 200 + fc];
    botv[j] = abot[fc] * w;
    part += cv * atop[fc] * w;
  }
  const float s_cur = warp_allreduce_sum(part);
  const float bias0 = ab[0];

  float nbv[6][7], sc6[6], am6[6];
#pragma unroll
  for (int k = 0; k < 6; ++k) {
    const int ia = idxA[(size_t)atom * 6 + k];
    const int ib = idxB[(size_t)atom * 6 + k];
    const bool pad = (ia == 127);
    am6[k] = pad ? 0.f : 1.f;
    const size_t baseA = ((size_t)(b * 128 + ia)) * 200;
    const size_t baseB = ((size_t)(b * 128 + ib)) * 200;
    float d = 0.f;
#pragma unroll
    for (int j = 0; j < 7; ++j) {
      const int f  = lane + 32 * j;
      const int fc = f < 200 ? f : 199;
      float v;
      if (first) v = f_lrelu(P[baseA + fc] + Qb[baseB + fc] + nbb[fc]);
      else       v = cur[baseA + fc];
      v = (f < 200) ? v : 0.f;
      nbv[k][j] = v;
      d += v * botv[j];
    }
    d = warp_allreduce_sum(d);
    sc6[k] = f_lrelu(s_cur + d + bias0) + (pad ? NEGC : 0.f);
  }
  float mx = sc6[0];
#pragma unroll
  for (int k = 1; k < 6; ++k) mx = fmaxf(mx, sc6[k]);
  float se = 0.f;
#pragma unroll
  for (int k = 0; k < 6; ++k) { sc6[k] = expf(sc6[k] - mx); se += sc6[k]; }
  const float inv = 1.f / se;
  float asv = 0.f;
#pragma unroll
  for (int k = 0; k < 6; ++k) { sc6[k] *= inv * am6[k]; asv += sc6[k]; }
#pragma unroll
  for (int j = 0; j < 7; ++j) {
    const int f = lane + 32 * j;
    if (f < 200) {
      float w = 0.f;
#pragma unroll
      for (int k = 0; k < 6; ++k) w += sc6[k] * nbv[k][j];
      wsum[(size_t)atom * 200 + f] = w;
    }
  }
  if (lane == 0) asum[atom] = asv;
}

// ---------------- GRUCell combine (gi/gh are [rows,600], biases added here) --
__global__ void k_gru_combine(const float* __restrict__ gi, const float* __restrict__ gh,
                              const float* __restrict__ bi, const float* __restrict__ bh,
                              const float* __restrict__ h_in, float* __restrict__ h_out,
                              float* __restrict__ cur_out, int rows)
{
  const int idx = blockIdx.x * blockDim.x + threadIdx.x;
  if (idx >= rows * 200) return;
  const int r = idx / 200, f = idx % 200;
  const size_t base = (size_t)r * 600;
  const float ir = gi[base + f]       + bi[f];
  const float iz = gi[base + 200 + f] + bi[200 + f];
  const float in = gi[base + 400 + f] + bi[400 + f];
  const float hr = gh[base + f]       + bh[f];
  const float hz = gh[base + 200 + f] + bh[200 + f];
  const float hn = gh[base + 400 + f] + bh[400 + f];
  const float rg = f_sigm(ir + hr);
  const float zg = f_sigm(iz + hz);
  const float ng = tanhf(in + rg * hn);
  const float hv = (1.f - zg) * ng + zg * h_in[idx];
  h_out[idx]   = hv;
  cur_out[idx] = f_relu(hv);
}

// ---------------- masked sum over L: mol_feature init ------------------------
__global__ void k_mask_sum(const float* __restrict__ cur, const float* __restrict__ mask,
                           float* __restrict__ mf, float* __restrict__ actm)
{
  const int idx = blockIdx.x * blockDim.x + threadIdx.x;
  if (idx >= 256 * 200) return;
  const int b = idx / 200, f = idx % 200;
  float s = 0.f;
  for (int l = 0; l < 128; ++l)
    s += cur[((size_t)(b * 128 + l)) * 200 + f] * mask[b * 128 + l];
  mf[idx] = s;
  actm[idx] = f_relu(s);
}

// ---------------- molecule-level attention (one block per molecule) ----------
__launch_bounds__(256)
__global__ void k_mol_attn(const float* __restrict__ actm, const float* __restrict__ cur,
                           const float* __restrict__ mtop, const float* __restrict__ mbot,
                           const float* __restrict__ mb, const float* __restrict__ mask,
                           float* __restrict__ wsm, float* __restrict__ asum)
{
  __shared__ float red[256];
  __shared__ float sc[128];
  const int b = blockIdx.x, tid = threadIdx.x;

  red[tid] = (tid < 200) ? actm[(size_t)b * 200 + tid] * mtop[tid] : 0.f;
  __syncthreads();
  for (int s = 128; s > 0; s >>= 1) { if (tid < s) red[tid] += red[tid + s]; __syncthreads(); }
  const float s_mol = red[0];
  __syncthreads();

  if (tid < 128) {
    const float* cp = &cur[((size_t)(b * 128 + tid)) * 200];
    float d = 0.f;
    for (int f = 0; f < 200; ++f) d += cp[f] * mbot[f];
    const float msk = mask[b * 128 + tid];
    sc[tid] = f_lrelu(s_mol + d + mb[0]) + (msk == 0.f ? NEGC : 0.f);
  }
  __syncthreads();

  red[tid] = (tid < 128) ? sc[tid] : -3.4e38f;
  __syncthreads();
  for (int s = 128; s > 0; s >>= 1) { if (tid < s) red[tid] = fmaxf(red[tid], red[tid + s]); __syncthreads(); }
  const float mx = red[0];
  __syncthreads();

  const float ex = (tid < 128) ? expf(sc[tid] - mx) : 0.f;
  red[tid] = ex;
  __syncthreads();
  for (int s = 128; s > 0; s >>= 1) { if (tid < s) red[tid] += red[tid + s]; __syncthreads(); }
  const float se = red[0];
  __syncthreads();

  if (tid < 128) sc[tid] = ex / se * mask[b * 128 + tid];
  __syncthreads();

  red[tid] = (tid < 128) ? sc[tid] : 0.f;
  __syncthreads();
  for (int s = 128; s > 0; s >>= 1) { if (tid < s) red[tid] += red[tid + s]; __syncthreads(); }
  if (tid == 0) asum[b] = red[0];

  if (tid < 200) {
    float w = 0.f;
    for (int l = 0; l < 128; ++l)
      w += sc[l] * cur[((size_t)(b * 128 + l)) * 200 + tid];
    wsm[(size_t)b * 200 + tid] = w;
  }
}

// ---------------- eval-mode BatchNorm1d (in place) ---------------------------
__global__ void k_bn(float* __restrict__ x, const float* __restrict__ g,
                     const float* __restrict__ b, const float* __restrict__ m,
                     const float* __restrict__ v, int M, int N)
{
  const int idx = blockIdx.x * blockDim.x + threadIdx.x;
  if (idx >= M * N) return;
  const int n = idx % N;
  x[idx] = (x[idx] - m[n]) * rsqrtf(v[n] + 1e-5f) * g[n] + b[n];
}

// ---------------- small dense (+relu)(+bn) for chem MLP ----------------------
__global__ void k_dense_small(const float* __restrict__ in, const float* __restrict__ w,
                              const float* __restrict__ b,
                              const float* __restrict__ bng, const float* __restrict__ bnb,
                              const float* __restrict__ bnm, const float* __restrict__ bnv,
                              float* __restrict__ out, int M, int K, int N, int act)
{
  const int idx = blockIdx.x * blockDim.x + threadIdx.x;
  if (idx >= M * N) return;
  const int m = idx / N, n = idx % N;
  float acc = b ? b[n] : 0.f;
  for (int k = 0; k < K; ++k) acc += in[(size_t)m * K + k] * w[(size_t)k * N + n];
  if (act == 1) acc = f_relu(acc);
  if (bng) acc = (acc - bnm[n]) * rsqrtf(bnv[n] + 1e-5f) * bng[n] + bnb[n];
  out[idx] = acc;
}

// ---------------- output head: logits + sigmoid ------------------------------
__global__ void k_output(const float* __restrict__ mpred, const float* __restrict__ c,
                         const float* __restrict__ ow, const float* __restrict__ ob,
                         float* __restrict__ out)
{
  const int b = blockIdx.x * blockDim.x + threadIdx.x;
  if (b >= 256) return;
  float acc = ob[0];
  for (int j = 0; j < 128; ++j) acc += mpred[(size_t)b * 128 + j] * ow[j];
  for (int j = 0; j < 20;  ++j) acc += c[(size_t)b * 20 + j] * ow[128 + j];
  out[b] = acc;
  out[256 + b] = f_sigm(acc);
}

// ============================================================================
static inline void gemm(hipStream_t s, const float* A, int lda, const _Float16* W,
                        const float* bias, const float* rowscale,
                        float* C, int ldc, float* C2, int M, int N, int K, int act)
{
  const int kpw = ((K + 31) / 32) * 32;
  dim3 g((N + 63) / 64, (M + 63) / 64);
  k_gemm<<<g, dim3(256), 0, s>>>(A, lda, W, kpw, bias, rowscale, C, ldc, C2, M, N, K, act);
}

static inline void convw(hipStream_t s, const float* src, _Float16* dst,
                         int K, int N, int Kpad, int Npad, int mode)
{
  int tot = Npad * Kpad;
  k_conv_w<<<dim3((tot + 255) / 256), dim3(256), 0, s>>>(src, dst, K, N, Kpad, Npad, mode);
}

extern "C" void kernel_launch(void* const* d_in, const int* in_sizes, int n_in,
                              void* d_out, int out_size, void* d_ws, size_t ws_size,
                              hipStream_t stream)
{
  (void)in_sizes; (void)n_in; (void)out_size; (void)ws_size;

  const float* x_atom = (const float*)d_in[0];   // [256,128,39]
  const float* x_bond = (const float*)d_in[1];   // [256,128,10]
  const int*   idxA   = (const int*)d_in[2];     // [256,128,6]
  const int*   idxB   = (const int*)d_in[3];
  const float* x_mask = (const float*)d_in[4];   // [256,128]
  const float* chem   = (const float*)d_in[5];   // [256,200]
  int pi = 6;
  const float* atom_fc_w     = (const float*)d_in[pi++];  // [39,200]
  const float* atom_fc_b     = (const float*)d_in[pi++];
  const float* nb_fc_w       = (const float*)d_in[pi++];  // [49,200]
  const float* nb_fc_b       = (const float*)d_in[pi++];
  const float* align_w       = (const float*)d_in[pi++];  // [3,400,1]
  const float* align_b       = (const float*)d_in[pi++];  // [3,1]
  const float* attend_w      = (const float*)d_in[pi++];  // [3,200,200]
  const float* attend_b      = (const float*)d_in[pi++];  // [3,200]
  const float* gru_wi        = (const float*)d_in[pi++];  // [3,600,200]
  const float* gru_wh        = (const float*)d_in[pi++];
  const float* gru_bi        = (const float*)d_in[pi++];  // [3,600]
  const float* gru_bh        = (const float*)d_in[pi++];
  const float* mol_align_w   = (const float*)d_in[pi++];  // [400,1]
  const float* mol_align_b   = (const float*)d_in[pi++];  // [1]
  const float* mol_attend_w  = (const float*)d_in[pi++];  // [200,200]
  const float* mol_attend_b  = (const float*)d_in[pi++];
  const float* mol_gru_wi    = (const float*)d_in[pi++];  // [600,200]
  const float* mol_gru_wh    = (const float*)d_in[pi++];
  const float* mol_gru_bi    = (const float*)d_in[pi++];  // [600]
  const float* mol_gru_bh    = (const float*)d_in[pi++];
  const float* mol_norm_g    = (const float*)d_in[pi++];  // [200]
  const float* mol_norm_b    = (const float*)d_in[pi++];
  const float* mol_norm_m    = (const float*)d_in[pi++];
  const float* mol_norm_v    = (const float*)d_in[pi++];
  const float* mol_output_w  = (const float*)d_in[pi++];  // [200,128]
  const float* mol_output_b  = (const float*)d_in[pi++];
  const float* fcw[5]; const float* fcb[5];
  for (int i = 0; i < 5; ++i) { fcw[i] = (const float*)d_in[pi++]; fcb[i] = (const float*)d_in[pi++]; }
  const float* bng[4]; const float* bnb[4]; const float* bnm[4]; const float* bnv[4];
  for (int i = 0; i < 4; ++i) {
    bng[i] = (const float*)d_in[pi++]; bnb[i] = (const float*)d_in[pi++];
    bnm[i] = (const float*)d_in[pi++]; bnv[i] = (const float*)d_in[pi++];
  }
  const float* output_w = (const float*)d_in[pi++];       // [148,1]
  const float* output_b = (const float*)d_in[pi++];

  // ---- workspace carve (bump allocator, 256B aligned) -----------------------
  char* wp = (char*)d_ws;
  auto take = [&](size_t bytes) -> void* {
    void* r = (void*)wp; wp += (bytes + 255) & ~(size_t)255; return r;
  };
  // f16 weights, N-major [Npad][Kpad], Npad % 64 == 0
  _Float16* Wafc  = (_Float16*)take((size_t)256 * 64 * 2);
  _Float16* Wnba  = (_Float16*)take((size_t)256 * 64 * 2);
  _Float16* Wnbb  = (_Float16*)take((size_t)256 * 32 * 2);
  _Float16* Watt[3]; _Float16* Wgi[3]; _Float16* Wgh[3];
  for (int r = 0; r < 3; ++r) {
    Watt[r] = (_Float16*)take((size_t)256 * 224 * 2);
    Wgi[r]  = (_Float16*)take((size_t)640 * 224 * 2);
    Wgh[r]  = (_Float16*)take((size_t)640 * 224 * 2);
  }
  _Float16* Wmatt = (_Float16*)take((size_t)256 * 224 * 2);
  _Float16* Wmgi  = (_Float16*)take((size_t)640 * 224 * 2);
  _Float16* Wmgh  = (_Float16*)take((size_t)640 * 224 * 2);
  _Float16* Wmout = (_Float16*)take((size_t)128 * 224 * 2);

  const size_t NA = 32768;                       // B*L atoms
  float* P    = (float*)take(NA * 200 * 4);
  float* Qb   = (float*)take(NA * 200 * 4);
  float* h    = (float*)take(NA * 200 * 4);
  float* cur  = (float*)take(NA * 200 * 4);
  float* wsum = (float*)take(NA * 200 * 4);
  float* ctx  = (float*)take(NA * 200 * 4);
  float* asum = (float*)take(NA * 4);
  float* giC  = (float*)take((size_t)8192 * 600 * 4);
  float* ghC  = (float*)take((size_t)8192 * 600 * 4);
  float* mf   = (float*)take((size_t)256 * 200 * 4);
  float* actm = (float*)take((size_t)256 * 200 * 4);
  float* wsm  = (float*)take((size_t)256 * 200 * 4);
  float* asmv = (float*)take((size_t)256 * 4);
  float* ctxm = (float*)take((size_t)256 * 200 * 4);
  float* mgi  = (float*)take((size_t)256 * 600 * 4);
  float* mgh  = (float*)take((size_t)256 * 600 * 4);
  float* mprd = (float*)take((size_t)256 * 128 * 4);
  float* cb1  = (float*)take((size_t)256 * 50 * 4);
  float* cb2  = (float*)take((size_t)256 * 50 * 4);

  // ---- convert weights to padded f16 (N-major) ------------------------------
  convw(stream, atom_fc_w, Wafc, 39, 200, 64, 256, 0);
  convw(stream, nb_fc_w,            Wnba, 39, 200, 64, 256, 0);   // rows 0..38
  convw(stream, nb_fc_w + 39 * 200, Wnbb, 10, 200, 32, 256, 0);   // rows 39..48
  for (int r = 0; r < 3; ++r) {
    convw(stream, attend_w + (size_t)r * 200 * 200, Watt[r], 200, 200, 224, 256, 0);
    convw(stream, gru_wi   + (size_t)r * 600 * 200, Wgi[r],  200, 600, 224, 640, 1);
    convw(stream, gru_wh   + (size_t)r * 600 * 200, Wgh[r],  200, 600, 224, 640, 1);
  }
  convw(stream, mol_attend_w, Wmatt, 200, 200, 224, 256, 0);
  convw(stream, mol_gru_wi,   Wmgi,  200, 600, 224, 640, 1);
  convw(stream, mol_gru_wh,   Wmgh,  200, 600, 224, 640, 1);
  convw(stream, mol_output_w, Wmout, 200, 128, 224, 128, 0);

  // ---- stage 1: atom_fc (h = cur = lrelu(x_atom@W+b)), factored neighbor_fc -
  gemm(stream, x_atom, 39, Wafc, atom_fc_b, nullptr, h,  200, cur,     32768, 200, 39, 1);
  gemm(stream, x_atom, 39, Wnba, nullptr,   nullptr, P,  200, nullptr, 32768, 200, 39, 0);
  gemm(stream, x_bond, 10, Wnbb, nullptr,   nullptr, Qb, 200, nullptr, 32768, 200, 10, 0);

  // ---- stage 2: R=3 graph-attention + GRU rounds ----------------------------
  for (int r = 0; r < 3; ++r) {
    k_neighbor<<<dim3(4096), dim3(256), 0, stream>>>(
        cur, P, Qb, nb_fc_b, idxA, idxB,
        align_w + (size_t)r * 400, align_w + (size_t)r * 400 + 200, align_b + r,
        wsum, asum, (r == 0) ? 1 : 0);
    // context = elu(wsum@attend_w + asum*attend_b)
    gemm(stream, wsum, 200, Watt[r], attend_b + (size_t)r * 200, asum,
         ctx, 200, nullptr, 32768, 200, 200, 3);
    // GRU in 8192-row chunks to bound workspace
    for (int c = 0; c < 4; ++c) {
      const size_t r0 = (size_t)c * 8192;
      gemm(stream, ctx + r0 * 200, 200, Wgi[r], nullptr, nullptr,
           giC, 600, nullptr, 8192, 600, 200, 0);
      gemm(stream, h   + r0 * 200, 200, Wgh[r], nullptr, nullptr,
           ghC, 600, nullptr, 8192, 600, 200, 0);
      k_gru_combine<<<dim3((8192 * 200 + 255) / 256), dim3(256), 0, stream>>>(
          giC, ghC, gru_bi + (size_t)r * 600, gru_bh + (size_t)r * 600,
          h + r0 * 200, h + r0 * 200, cur + r0 * 200, 8192);
    }
  }

  // ---- stage 3: molecule readout (T=2) --------------------------------------
  k_mask_sum<<<dim3(200), dim3(256), 0, stream>>>(cur, x_mask, mf, actm);
  for (int t = 0; t < 2; ++t) {
    k_mol_attn<<<dim3(256), dim3(256), 0, stream>>>(
        actm, cur, mol_align_w, mol_align_w + 200, mol_align_b, x_mask, wsm, asmv);
    gemm(stream, wsm, 200, Wmatt, mol_attend_b, asmv, ctxm, 200, nullptr, 256, 200, 200, 3);
    k_bn<<<dim3(200), dim3(256), 0, stream>>>(ctxm, mol_norm_g, mol_norm_b,
                                              mol_norm_m, mol_norm_v, 256, 200);
    gemm(stream, ctxm, 200, Wmgi, nullptr, nullptr, mgi, 600, nullptr, 256, 600, 200, 0);
    gemm(stream, mf,   200, Wmgh, nullptr, nullptr, mgh, 600, nullptr, 256, 600, 200, 0);
    k_gru_combine<<<dim3((256 * 200 + 255) / 256), dim3(256), 0, stream>>>(
        mgi, mgh, mol_gru_bi, mol_gru_bh, mf, mf, actm, 256);
  }
  gemm(stream, mf, 200, Wmout, mol_output_b, nullptr, mprd, 128, nullptr, 256, 128, 200, 0);

  // ---- stage 4: chemical-info MLP -------------------------------------------
  const int dims[6] = {200, 50, 40, 32, 20, 20};
  k_dense_small<<<dim3((256 * 50 + 255) / 256), dim3(256), 0, stream>>>(
      chem, fcw[0], fcb[0], bng[0], bnb[0], bnm[0], bnv[0], cb1, 256, dims[0], dims[1], 1);
  k_dense_small<<<dim3((256 * 40 + 255) / 256), dim3(256), 0, stream>>>(
      cb1, fcw[1], fcb[1], bng[1], bnb[1], bnm[1], bnv[1], cb2, 256, dims[1], dims[2], 1);
  k_dense_small<<<dim3((256 * 32 + 255) / 256), dim3(256), 0, stream>>>(
      cb2, fcw[2], fcb[2], bng[2], bnb[2], bnm[2], bnv[2], cb1, 256, dims[2], dims[3], 1);
  k_dense_small<<<dim3((256 * 20 + 255) / 256), dim3(256), 0, stream>>>(
      cb1, fcw[3], fcb[3], bng[3], bnb[3], bnm[3], bnv[3], cb2, 256, dims[3], dims[4], 1);
  k_dense_small<<<dim3((256 * 20 + 255) / 256), dim3(256), 0, stream>>>(
      cb2, fcw[4], fcb[4], nullptr, nullptr, nullptr, nullptr, cb1, 256, dims[4], dims[5], 0);

  // ---- stage 5: output head: [logits | sigmoid(logits)] ---------------------
  k_output<<<dim3(1), dim3(256), 0, stream>>>(mprd, cb1, output_w, output_b, (float*)d_out);
}
